// DMTetGeometry_5222680232581
// MI455X (gfx1250) — compile-verified, matrix operand order checked by reading, not run on hardware
//
#include <hip/hip_runtime.h>
#include <stdint.h>

// ---------------- problem constants (grid_res = 64, fixed by reference) ----
#define RES      64
#define R1       65
#define NV       (R1*R1*R1)          // 274625 grid vertices
#define NV3      (NV*3)
#define NT       (6*RES*RES*RES)     // 1572864 tets
#define ES       (NV*7)              // 1922375 canonical edge slots
#define BLK      8192                // elements per workgroup (8 waves x 1024)
#define NB_E     ((ES + BLK - 1)/BLK) // 235
#define NB_T     (NT/BLK)             // 192
#define NN       1255                // ceil(sqrt((2*NT+1)//2))
#define SDF_TILE 5504                // 1171 vid span + 4291 max delta + pad

typedef __attribute__((ext_vector_type(8))) int v8i;

// triangle table (edge local ids 0..5, -1 = unused)
__constant__ signed char TRI_TAB[16][6] = {
  {-1,-1,-1,-1,-1,-1},{ 1, 0, 2,-1,-1,-1},{ 4, 0, 3,-1,-1,-1},{ 1, 4, 2, 1, 3, 4},
  { 3, 1, 5,-1,-1,-1},{ 2, 3, 0, 2, 5, 3},{ 1, 4, 0, 1, 5, 4},{ 4, 2, 5,-1,-1,-1},
  { 4, 5, 2,-1,-1,-1},{ 4, 1, 0, 4, 5, 1},{ 3, 2, 0, 3, 5, 2},{ 1, 3, 5,-1,-1,-1},
  { 4, 1, 2, 4, 3, 1},{ 3, 0, 4,-1,-1,-1},{ 2, 0, 1,-1,-1,-1},{-1,-1,-1,-1,-1,-1}
};

// ---- CDNA5: iu8 WMMA as a 1024-way integer reduction (D = A x ones + 0) ----
// A: 16x64 u8 counts striped over the wave (32 bytes / lane). With B = ones,
// D[m][n] = rowsum_m (exact i32). Lane 0 holds rows 0-7 at N=0, lane 16 rows
// 8-15, so wave total = readlane(s,0) + readlane(s,16).
__device__ inline int wmma_sum_u8(v8i a) {
  v8i ones = {0x01010101,0x01010101,0x01010101,0x01010101,
              0x01010101,0x01010101,0x01010101,0x01010101};
  v8i c    = {0,0,0,0,0,0,0,0};
  v8i d = __builtin_amdgcn_wmma_i32_16x16x64_iu8(false, a, false, ones, c, false, false);
  int s = d[0]+d[1]+d[2]+d[3]+d[4]+d[5]+d[6]+d[7];
  return __builtin_amdgcn_readlane(s, 0) + __builtin_amdgcn_readlane(s, 16);
}

// ---- CDNA5: async global->LDS staging of the block's SDF window ------------
// Low 32 bits of a generic __shared__ pointer == wave-relative LDS byte addr.
__device__ inline void stage_sdf(float* s_sdf, const float* __restrict__ sdf, int vid_base) {
  for (int i = (int)threadIdx.x; i < SDF_TILE; i += 256) {
    int g = vid_base + i; if (g >= NV) g = NV - 1;   // clamp: duplicates are fine
    unsigned           lds = (unsigned)(unsigned long long)(&s_sdf[i]);
    unsigned long long ga  = (unsigned long long)(sdf + g);
    asm volatile("global_load_async_to_lds_b32 %0, %1, off"
                 :: "v"(lds), "v"(ga) : "memory");
  }
  asm volatile("s_wait_asynccnt 0" ::: "memory");
  __syncthreads();
}

// canonical edge slot -> (flag, vid, delta).  slot = vid*7 + r, r encodes
// direction bits m = r+1 = (di,dj,dk); vid delta = di*4225 + dj*65 + dk.
struct EdgeInfo { int flag, vid, dd; };
__device__ inline EdgeInfo edge_info(int s, const float* s_sdf, int vid_base) {
  EdgeInfo e; e.flag = 0; e.vid = 0; e.dd = 0;
  if (s >= ES) return e;
  int vid = s / 7; int r = s - vid*7; int m = r + 1;
  int di = (m>>2)&1, dj = (m>>1)&1, dk = m&1;
  int i = vid/4225; int rem = vid - i*4225; int j = rem/65; int k = rem - j*65;
  if (i+di > RES || j+dj > RES || k+dk > RES) return e;   // off-grid edge
  int dd = di*4225 + dj*65 + dk;
  float a = s_sdf[vid - vid_base];
  float b = s_sdf[vid - vid_base + dd];
  e.vid = vid; e.dd = dd;
  e.flag = ((a > 0.f) != (b > 0.f)) ? 1 : 0;
  return e;
}

__device__ inline int tet_occ(const int4 vv, const float* __restrict__ sdf) {
  return (sdf[vv.x] > 0.f ? 1:0) | (sdf[vv.y] > 0.f ? 2:0) |
         (sdf[vv.z] > 0.f ? 4:0) | (sdf[vv.w] > 0.f ? 8:0);
}
__device__ inline int ntri_of(int occ) {
  int pc = __popc(occ);
  return (pc == 2) ? 2 : ((pc == 1 || pc == 3) ? 1 : 0);
}

// K0: v_deformed = verts + (1/64) * tanh(deform)
__global__ __launch_bounds__(256) void k_deform(const float* __restrict__ verts,
    const float* __restrict__ deform, float* __restrict__ vdef) {
  int i = blockIdx.x*256 + threadIdx.x;
  if (i < NV3) vdef[i] = verts[i] + (1.0f/64.0f)*tanhf(deform[i]);
}

// K1: per-block count of sign-crossing edges (async SDF stage + iu8 WMMA)
__global__ __launch_bounds__(256) void k_edge_count(const float* __restrict__ sdf,
                                                    int* __restrict__ cntE) {
  __shared__ float s_sdf[SDF_TILE];
  __shared__ int   s_w[8];
  int b = blockIdx.x; int base = b*BLK; int vid_base = base/7;
  stage_sdf(s_sdf, sdf, vid_base);
  int wave = threadIdx.x >> 5, lane = threadIdx.x & 31;
  int s0 = base + wave*1024 + lane*32;
  v8i a;
  #pragma unroll
  for (int d = 0; d < 8; ++d) {
    int pk = 0;
    #pragma unroll
    for (int q = 0; q < 4; ++q)
      pk |= edge_info(s0 + d*4 + q, s_sdf, vid_base).flag << (8*q);
    a[d] = pk;
  }
  int tot = wmma_sum_u8(a);            // 1024 flags summed by one WMMA
  if (lane == 0) s_w[wave] = tot;
  __syncthreads();
  if (threadIdx.x == 0) { int t = 0; for (int w = 0; w < 8; ++w) t += s_w[w]; cntE[b] = t; }
}

// K3: per-block (num_tri==1, num_tri==2) counts via two iu8 WMMAs
__global__ __launch_bounds__(256) void k_tet_count(const int* __restrict__ tets,
    const float* __restrict__ sdf, int* __restrict__ cnt1, int* __restrict__ cnt2) {
  __shared__ int sw1[8], sw2[8];
  const int4* t4 = (const int4*)tets;
  int b = blockIdx.x;
  int wave = threadIdx.x >> 5, lane = threadIdx.x & 31;
  int tb = b*BLK + wave*1024 + lane*32;
  v8i a1, a2;
  #pragma unroll
  for (int d = 0; d < 8; ++d) {
    int p1 = 0, p2 = 0;
    #pragma unroll
    for (int q = 0; q < 4; ++q) {
      int nt = ntri_of(tet_occ(t4[tb + d*4 + q], sdf));
      p1 |= (nt == 1 ? 1 : 0) << (8*q);
      p2 |= (nt == 2 ? 1 : 0) << (8*q);
    }
    a1[d] = p1; a2[d] = p2;
  }
  int t1 = wmma_sum_u8(a1);
  int t2 = wmma_sum_u8(a2);
  if (lane == 0) { sw1[wave] = t1; sw2[wave] = t2; }
  __syncthreads();
  if (threadIdx.x == 0) {
    int s1 = 0, s2 = 0;
    for (int w = 0; w < 8; ++w) { s1 += sw1[w]; s2 += sw2[w]; }
    cnt1[b] = s1; cnt2[b] = s2;
  }
}

// K2: exclusive scans of the (<=256) per-block counts; totals to ws
__device__ inline void scan_one(const int* __restrict__ in, int n,
    int* __restrict__ out_excl, int* __restrict__ total, int* s) {
  int t = threadIdx.x;
  int v = (t < n) ? in[t] : 0;
  s[t] = v; __syncthreads();
  for (int o = 1; o < 256; o <<= 1) {
    int u = (t >= o) ? s[t - o] : 0;
    __syncthreads();
    s[t] += u;
    __syncthreads();
  }
  if (t < n) out_excl[t] = s[t] - v;
  if (t == n - 1) *total = s[t];
  __syncthreads();
}
__global__ __launch_bounds__(256) void k_scan(const int* cntE, int* offE, int nE,
    const int* cnt1, int* off1, const int* cnt2, int* off2, int nT, int* totals) {
  __shared__ int s[256];
  scan_one(cntE, nE, offE, &totals[0], s);
  scan_one(cnt1, nT, off1, &totals[1], s);
  scan_one(cnt2, nT, off2, &totals[2], s);
}

// K4: emit interpolated vertices (slot order == jnp.unique order) + slot->idx map
__global__ __launch_bounds__(256) void k_edge_emit(const float* __restrict__ sdf,
    const float* __restrict__ vdef, const int* __restrict__ offE,
    int* __restrict__ mapping, float* __restrict__ outf) {
  __shared__ float s_sdf[SDF_TILE];
  __shared__ int   s_sc[256];
  int b = blockIdx.x; int base = b*BLK; int vid_base = base/7;
  stage_sdf(s_sdf, sdf, vid_base);
  int t = threadIdx.x;
  int s0 = base + t*32;                     // contiguous -> scan order == slot order
  unsigned flags = 0; int cnt = 0;
  for (int q = 0; q < 32; ++q) {
    EdgeInfo e = edge_info(s0 + q, s_sdf, vid_base);
    flags |= ((unsigned)e.flag) << q;
    cnt += e.flag;
  }
  int v = cnt; s_sc[t] = v; __syncthreads();
  for (int o = 1; o < 256; o <<= 1) {
    int u = (t >= o) ? s_sc[t - o] : 0;
    __syncthreads(); s_sc[t] += u; __syncthreads();
  }
  int gi = offE[b] + (s_sc[t] - v);
  for (int q = 0; q < 32; ++q) {
    if ((flags >> q) & 1u) {
      EdgeInfo e = edge_info(s0 + q, s_sdf, vid_base);
      float sa = s_sdf[e.vid - vid_base];
      float sb = s_sdf[e.vid - vid_base + e.dd];
      float denom = sa - sb;
      float w0 = -sb/denom, w1 = sa/denom;     // p0*(-b/(a-b)) + p1*(a/(a-b))
      int v0 = e.vid, v1 = e.vid + e.dd;
      mapping[s0 + q] = gi;
      outf[3*gi+0] = vdef[3*v0+0]*w0 + vdef[3*v1+0]*w1;
      outf[3*gi+1] = vdef[3*v0+1]*w0 + vdef[3*v1+1]*w1;
      outf[3*gi+2] = vdef[3*v0+2]*w0 + vdef[3*v1+2]*w1;
      ++gi;
    }
  }
}

// K5: emit faces (1-tri tets first, then 2-tri tets) + uv_idx
__global__ __launch_bounds__(256) void k_tet_emit(const int* __restrict__ tets,
    const float* __restrict__ sdf, const int* __restrict__ mapping,
    const int* __restrict__ off1, const int* __restrict__ off2,
    const int* __restrict__ totals, int* __restrict__ outi) {
  __shared__ int s_sc[256];
  const int4* t4 = (const int4*)tets;
  int b = blockIdx.x, t = threadIdx.x;
  int base = b*BLK + t*32;
  int c1 = 0, c2 = 0;
  for (int q = 0; q < 32; ++q) {
    int nt = ntri_of(tet_occ(t4[base + q], sdf));
    c1 += (nt == 1); c2 += (nt == 2);
  }
  int v = c1 | (c2 << 16);
  s_sc[t] = v; __syncthreads();
  for (int o = 1; o < 256; o <<= 1) {
    int u = (t >= o) ? s_sc[t - o] : 0;
    __syncthreads(); s_sc[t] += u; __syncthreads();
  }
  int excl = s_sc[t] - v;
  int r1 = off1[b] + (excl & 0xffff);
  int r2 = off2[b] + (excl >> 16);
  int totE = totals[0], tot1 = totals[1], tot2 = totals[2];
  long long fbase = 3LL*totE;                                       // faces after verts
  long long ubase = fbase + 3LL*(tot1 + 2LL*tot2) + 8LL*NN*NN;      // uv_idx after uvs
  for (int q = 0; q < 32; ++q) {
    int g = base + q;
    int4 vv = t4[g];
    int occ = tet_occ(vv, sdf);
    int nt  = ntri_of(occ);
    if (!nt) continue;
    int vsv[4] = {vv.x, vv.y, vv.z, vv.w};
    const int pa[6] = {0,0,0,1,1,2};
    const int pb[6] = {1,2,3,2,3,3};
    int vmap[6];
    #pragma unroll
    for (int e = 0; e < 6; ++e) {
      int va = vsv[pa[e]], vb = vsv[pb[e]];
      int lo = va < vb ? va : vb;
      int hi = va < vb ? vb : va;
      int d  = hi - lo;
      int di = d/4225; int rem = d - di*4225; int dj = rem/65; int dk = rem - dj*65;
      int m  = (di<<2) | (dj<<1) | dk;          // rank = m-1
      vmap[e] = mapping[lo*7 + (m - 1)];
    }
    const signed char* row = TRI_TAB[occ];
    if (nt == 1) {
      long long fr = r1;
      outi[fbase+3*fr+0] = vmap[(int)row[0]];
      outi[fbase+3*fr+1] = vmap[(int)row[1]];
      outi[fbase+3*fr+2] = vmap[(int)row[2]];
      outi[ubase+3*fr+0] = 4*g; outi[ubase+3*fr+1] = 4*g+1; outi[ubase+3*fr+2] = 4*g+2;
      ++r1;
    } else {
      long long fr = tot1 + 2LL*r2;
      outi[fbase+3*fr+0] = vmap[(int)row[0]];
      outi[fbase+3*fr+1] = vmap[(int)row[1]];
      outi[fbase+3*fr+2] = vmap[(int)row[2]];
      outi[ubase+3*fr+0] = 4*g; outi[ubase+3*fr+1] = 4*g+1; outi[ubase+3*fr+2] = 4*g+2;
      long long f2 = fr + 1;
      outi[fbase+3*f2+0] = vmap[(int)row[3]];
      outi[fbase+3*f2+1] = vmap[(int)row[4]];
      outi[fbase+3*f2+2] = vmap[(int)row[5]];
      outi[ubase+3*f2+0] = 4*g; outi[ubase+3*f2+1] = 4*g+2; outi[ubase+3*f2+2] = 4*g+3;
      ++r2;
    }
  }
}

// K6: uv grid (NN x NN cells, 4 uv rows each)
__global__ __launch_bounds__(256) void k_uvs(const int* __restrict__ totals,
                                             float* __restrict__ outf) {
  int idx = blockIdx.x*256 + threadIdx.x;
  if (idx >= NN*NN) return;
  int totE = totals[0], tot1 = totals[1], tot2 = totals[2];
  long long ub = 3LL*totE + 3LL*(tot1 + 2LL*tot2);
  int r = idx / NN, c = idx - r*NN;
  float inv = 1.0f/(float)NN;
  float x = c*inv, y = r*inv, pd = 0.9f*inv;
  float* o = outf + ub + (long long)idx*8;
  o[0]=x;    o[1]=y;
  o[2]=x+pd; o[3]=y;
  o[4]=x+pd; o[5]=y+pd;
  o[6]=x;    o[7]=y+pd;
}

extern "C" void kernel_launch(void* const* d_in, const int* in_sizes, int n_in,
                              void* d_out, int out_size, void* d_ws, size_t ws_size,
                              hipStream_t stream) {
  const float* verts  = (const float*)d_in[0];
  const float* sdf    = (const float*)d_in[1];
  const float* deform = (const float*)d_in[2];
  const int*   tets   = (const int*)d_in[3];
  (void)in_sizes; (void)n_in; (void)out_size; (void)ws_size;

  char* p = (char*)d_ws;
  float* vdef    = (float*)p; p += (((size_t)NV3*4) + 255) & ~(size_t)255;
  int*   mapping = (int*)p;   p += (((size_t)ES*4)  + 255) & ~(size_t)255;
  int* cntE = (int*)p; p += 1024;
  int* offE = (int*)p; p += 1024;
  int* cnt1 = (int*)p; p += 1024;
  int* off1 = (int*)p; p += 1024;
  int* cnt2 = (int*)p; p += 1024;
  int* off2 = (int*)p; p += 1024;
  int* totals = (int*)p;

  float* outf = (float*)d_out;
  int*   outi = (int*)d_out;

  k_deform    <<<(NV3 + 255)/256, 256, 0, stream>>>(verts, deform, vdef);
  k_edge_count<<<NB_E,            256, 0, stream>>>(sdf, cntE);
  k_tet_count <<<NB_T,            256, 0, stream>>>(tets, sdf, cnt1, cnt2);
  k_scan      <<<1,               256, 0, stream>>>(cntE, offE, NB_E, cnt1, off1, cnt2, off2, NB_T, totals);
  k_edge_emit <<<NB_E,            256, 0, stream>>>(sdf, vdef, offE, mapping, outf);
  k_tet_emit  <<<NB_T,            256, 0, stream>>>(tets, sdf, mapping, off1, off2, totals, outi);
  k_uvs       <<<(NN*NN + 255)/256, 256, 0, stream>>>(totals, outf);
}
